// Attention_81801947120105
// MI455X (gfx1250) — compile-verified, matrix operand order checked by reading, not run on hardware
//
#include <hip/hip_runtime.h>

typedef __attribute__((ext_vector_type(16))) _Float16 v16h;
typedef __attribute__((ext_vector_type(8)))  _Float16 v8h;
typedef __attribute__((ext_vector_type(2)))  __fp16   v2hf;  // cvt_pkrtz return type
typedef __attribute__((ext_vector_type(8)))  float    v8f;

#define B_    2
#define S_    2048
#define H_    32
#define HKV_  8
#define D_    128
#define G_    4
#define QTILE 128   // q rows per workgroup (8 waves x 16)
#define KBLK  64    // keys per staged block (2 x 32-key WMMA chunks)
#define KVSTR (HKV_ * D_)   // 1024 floats between consecutive tokens in k/v

__launch_bounds__(256, 2)
__global__ void fa_causal_gqa_wmma(const float* __restrict__ q,
                                   const float* __restrict__ k,
                                   const float* __restrict__ v,
                                   float* __restrict__ out)
{
    __shared__ _Float16 Ks[KBLK][D_];   // [key][dim]            16 KB
    __shared__ _Float16 Vt[D_][KBLK];   // [dim][key] transposed 16 KB

    const int tid  = threadIdx.x;
    const int lane = tid & 31;
    const int wv   = tid >> 5;
    const int ln16 = lane & 15;
    const int hi   = lane >> 4;          // lane half: 0 or 1

    const int QB = S_ / QTILE;           // 16
    int wg = blockIdx.x;
    const int qb  = wg % QB;  wg /= QB;
    const int hm  = wg % H_;  wg /= H_;  // query head (memory order: hkv*G + g)
    const int b   = wg;
    const int hkv = hm / G_;

    // SCALE in exp2 domain: p = exp2(raw * scale2 - m)
    const float scale2 = 0.08838834764831845f * 1.44269504088896340736f;

    const float* qP = q   + ((size_t)b * S_) * (H_ * D_) + (size_t)hm  * D_;
    const float* kP = k   + ((size_t)b * S_) * KVSTR     + (size_t)hkv * D_;
    const float* vP = v   + ((size_t)b * S_) * KVSTR     + (size_t)hkv * D_;
    float*       oP = out + ((size_t)b * S_) * (H_ * D_) + (size_t)hm  * D_;

    const int qRow0 = qb * QTILE + wv * 16;  // first q row of this wave's tile
    const int qPos  = qRow0 + ln16;          // this lane's q column

    // Q^T B-fragments (32x16 f16): B[lane,h] = Q[qPos][32*dblk + kmap(h)]
    v16h qf[4];
    #pragma unroll
    for (int dblk = 0; dblk < 4; ++dblk) {
        const float* qrow = qP + (size_t)qPos * (H_ * D_) + dblk * 32 + hi * 8;
        #pragma unroll
        for (int h2 = 0; h2 < 8; ++h2) {
            qf[dblk][h2]     = (_Float16)qrow[h2];
            qf[dblk][h2 + 8] = (_Float16)qrow[16 + h2];
        }
    }

    v8f acc[8];                          // O^T: 8 tiles of 16 dims x 16 q cols
    #pragma unroll
    for (int t = 0; t < 8; ++t) acc[t] = v8f{};
    float m_run = -1e30f;                // running max (scaled domain)
    float l_run = 0.0f;

    const int numBlocks = (qb + 1) * (QTILE / KBLK);  // keys < qb*128 + 128

    for (int kb = 0; kb < numBlocks; ++kb) {
        const int kBase = kb * KBLK;
        __syncthreads();

        // ---- stage K: 1024 float8 chunks, 4 per thread -> ds_store_b128 ----
        #pragma unroll
        for (int p = 0; p < 4; ++p) {
            int u  = tid + p * 256;
            int kk = u >> 4;                 // key 0..63
            int d8 = (u & 15) * 8;           // dim chunk
            const float4* src =
                (const float4*)(kP + (size_t)(kBase + kk) * KVSTR + d8);
            float4 a = src[0], c = src[1];
            v8h hv;
            ((v2hf*)&hv)[0] = __builtin_amdgcn_cvt_pkrtz(a.x, a.y);
            ((v2hf*)&hv)[1] = __builtin_amdgcn_cvt_pkrtz(a.z, a.w);
            ((v2hf*)&hv)[2] = __builtin_amdgcn_cvt_pkrtz(c.x, c.y);
            ((v2hf*)&hv)[3] = __builtin_amdgcn_cvt_pkrtz(c.z, c.w);
            *(v8h*)&Ks[kk][d8] = hv;
        }
        // ---- stage V transposed: float2 loads cover two dims at once ----
        #pragma unroll
        for (int p = 0; p < 2; ++p) {
            int task = tid + p * 256;        // 512 tasks: 64 dim-pairs x 8 kgrps
            int dd2  = (task & 63) * 2;      // even dim
            int kk0  = (task >> 6) * 8;      // key group 0..56
            float2 f[8];
            #pragma unroll
            for (int j = 0; j < 8; ++j)
                f[j] = *(const float2*)(vP + (size_t)(kBase + kk0 + j) * KVSTR + dd2);
            v8h h0, h1;
            #pragma unroll
            for (int jj = 0; jj < 4; ++jj) {
                ((v2hf*)&h0)[jj] = __builtin_amdgcn_cvt_pkrtz(f[2*jj].x, f[2*jj+1].x);
                ((v2hf*)&h1)[jj] = __builtin_amdgcn_cvt_pkrtz(f[2*jj].y, f[2*jj+1].y);
            }
            *(v8h*)&Vt[dd2][kk0]     = h0;
            *(v8h*)&Vt[dd2 + 1][kk0] = h1;
        }

        // ---- prefetch next block's K/V lines (gfx1250 global_prefetch) ----
        if (kb + 1 < numBlocks) {
            int pkk = tid >> 2;              // key 0..63
            int pdd = (tid & 3) * 32;        // 128B chunk within row
            __builtin_prefetch(kP + (size_t)(kBase + KBLK + pkk) * KVSTR + pdd, 0, 3);
            __builtin_prefetch(vP + (size_t)(kBase + KBLK + pkk) * KVSTR + pdd, 0, 3);
        }
        __syncthreads();

        if (kBase <= qRow0 + 15) {   // wave-uniform: EXEC stays all-ones
            // ---- S^T = K x Q^T : four 16-key tiles, accumulate over D ----
            v8f st[4] = { v8f{}, v8f{}, v8f{}, v8f{} };
            #pragma unroll
            for (int dblk = 0; dblk < 4; ++dblk) {
                const int dbase = dblk * 32 + hi * 8;
                #pragma unroll
                for (int tt = 0; tt < 4; ++tt) {
                    v16h a;
                    #pragma unroll
                    for (int h2 = 0; h2 < 8; ++h2) {
                        a[h2]     = Ks[tt * 16 + ln16][dbase + h2];
                        a[h2 + 8] = Ks[tt * 16 + ln16][dbase + 16 + h2];
                    }
                    st[tt] = __builtin_amdgcn_wmma_f32_16x16x32_f16(
                                 false, a, false, qf[dblk], (short)0, st[tt],
                                 false, false);
                }
            }

            // ---- causal mask only on diagonal blocks (wave-uniform) ----
            if (kBase + (KBLK - 1) > qRow0) {
                #pragma unroll
                for (int tt = 0; tt < 4; ++tt) {
                    #pragma unroll
                    for (int r = 0; r < 8; ++r) {
                        int key = kBase + tt * 16 + hi * 8 + r;
                        if (key > qPos) st[tt][r] = -3.0e38f;
                    }
                }
            }

            // ---- block max (raw), scale once ----
            float bmr = -3.0e38f;
            #pragma unroll
            for (int tt = 0; tt < 4; ++tt)
                #pragma unroll
                for (int r = 0; r < 8; ++r)
                    bmr = fmaxf(bmr, st[tt][r]);
            bmr = fmaxf(bmr, __shfl_xor(bmr, 16, 32));   // join lane pair
            const float m_new = fmaxf(m_run, bmr * scale2);
            // raw v_exp_f32: args are <= 0; deep-negative flushes to 0 (wanted)
            const float alpha = __builtin_amdgcn_exp2f(m_run - m_new);

            // ---- probs + two P^T B-fragments (keys 0-31 / 32-63, in-register) ----
            float rs = 0.0f;
            v16h pbA, pbB;
            #pragma unroll
            for (int r = 0; r < 8; ++r) {
                float p0 = __builtin_amdgcn_exp2f(fmaf(st[0][r], scale2, -m_new));
                float p1 = __builtin_amdgcn_exp2f(fmaf(st[1][r], scale2, -m_new));
                float p2 = __builtin_amdgcn_exp2f(fmaf(st[2][r], scale2, -m_new));
                float p3 = __builtin_amdgcn_exp2f(fmaf(st[3][r], scale2, -m_new));
                rs += (p0 + p1) + (p2 + p3);
                pbA[r] = (_Float16)p0;  pbA[r + 8] = (_Float16)p1;
                pbB[r] = (_Float16)p2;  pbB[r + 8] = (_Float16)p3;
            }
            rs += __shfl_xor(rs, 16, 32);
            l_run = l_run * alpha + rs;
            m_run = m_new;

            // ---- rescale O^T once per 64 keys, then O^T += V^T x P^T ----
            #pragma unroll
            for (int t = 0; t < 8; ++t) {
                #pragma unroll
                for (int r = 0; r < 8; ++r) acc[t][r] *= alpha;
            }
            #pragma unroll
            for (int t = 0; t < 8; ++t) {
                const int dimRow = t * 16 + ln16;
                const int kb2 = hi * 8;
                v16h av0, av1;
                #pragma unroll
                for (int h2 = 0; h2 < 8; ++h2) {
                    av0[h2]     = Vt[dimRow][kb2 + h2];
                    av0[h2 + 8] = Vt[dimRow][kb2 + 16 + h2];
                    av1[h2]     = Vt[dimRow][32 + kb2 + h2];
                    av1[h2 + 8] = Vt[dimRow][32 + kb2 + 16 + h2];
                }
                acc[t] = __builtin_amdgcn_wmma_f32_16x16x32_f16(
                             false, av0, false, pbA, (short)0, acc[t], false, false);
                acc[t] = __builtin_amdgcn_wmma_f32_16x16x32_f16(
                             false, av1, false, pbB, (short)0, acc[t], false, false);
            }
        }
    }

    // ---- finalize: divide by l, write O (un-transpose via index math) ----
    const float inv = 1.0f / l_run;
    #pragma unroll
    for (int t = 0; t < 8; ++t) {
        float* orow = oP + (size_t)qPos * (H_ * D_) + t * 16 + hi * 8;
        float4 o0 = { acc[t][0] * inv, acc[t][1] * inv,
                      acc[t][2] * inv, acc[t][3] * inv };
        float4 o1 = { acc[t][4] * inv, acc[t][5] * inv,
                      acc[t][6] * inv, acc[t][7] * inv };
        ((float4*)orow)[0] = o0;
        ((float4*)orow)[1] = o1;
    }
}

extern "C" void kernel_launch(void* const* d_in, const int* in_sizes, int n_in,
                              void* d_out, int out_size, void* d_ws, size_t ws_size,
                              hipStream_t stream) {
    (void)in_sizes; (void)n_in; (void)out_size; (void)d_ws; (void)ws_size;
    const float* q = (const float*)d_in[0];
    const float* k = (const float*)d_in[1];
    const float* v = (const float*)d_in[2];
    float* out = (float*)d_out;

    const int QB = S_ / QTILE;                // 16
    dim3 grid(B_ * H_ * QB);                  // 1024 workgroups
    dim3 block(256);                          // 8 wave32
    fa_causal_gqa_wmma<<<grid, block, 0, stream>>>(q, k, v, out);
}
